// MultiHeadAttention_13838384627961
// MI455X (gfx1250) — matrix operand order repaired and checked by reading, not compiled, not run on hardware
//
#include <hip/hip_runtime.h>
#include <hip/hip_bf16.h>
#include <stdint.h>

// ---- problem constants ----
#define BB   8
#define TT   2048
#define CC   512
#define HH   8
#define DD   64
#define MROWS (BB*TT)          // 16384
#define N_QKV (3*CC)           // 1536

typedef __attribute__((ext_vector_type(16))) __bf16 v16bf;
typedef __attribute__((ext_vector_type(8)))  __bf16 v8bf;
typedef __attribute__((ext_vector_type(8)))  float  v8f;

#define NEG_INF (-__builtin_inff())

#if defined(__AMDGCN__) && __has_builtin(__builtin_amdgcn_tensor_load_to_lds)
#define USE_TDM 1
#else
#define USE_TDM 0
#endif

#if USE_TDM
#if __has_builtin(__builtin_amdgcn_s_wait_tensorcnt)
#define WAIT_COPY() __builtin_amdgcn_s_wait_tensorcnt(0)
#else
#define WAIT_COPY() asm volatile("s_wait_tensorcnt 0" ::: "memory")
#endif
#else
// sync fallback: make this wave's ds_stores visible before the barrier
#define WAIT_COPY() asm volatile("s_wait_dscnt 0" ::: "memory")
#endif

static __device__ __forceinline__ unsigned short f2bf(float f) {
    __bf16 h = (__bf16)f;
    return __builtin_bit_cast(unsigned short, h);
}

// Build one 16-element bf16 fragment from two 16-byte-aligned 8-element runs.
static __device__ __forceinline__ v16bf make_frag(const unsigned short* lo,
                                                  const unsigned short* hi) {
    v8bf a = *(const v8bf*)lo;
    v8bf b = *(const v8bf*)hi;
    return __builtin_shufflevector(a, b, 0,1,2,3,4,5,6,7,8,9,10,11,12,13,14,15);
}

static __device__ __forceinline__ v8f zero8() {
    v8f z = {0.f,0.f,0.f,0.f,0.f,0.f,0.f,0.f};
    return z;
}

// generic pointer to an LDS object carries the LDS byte offset in bits [31:0]
static __device__ __forceinline__ unsigned lds_off(const void* p) {
    return (unsigned)(unsigned long long)p;
}

#if USE_TDM
typedef __attribute__((ext_vector_type(4))) unsigned int u32x4;
typedef __attribute__((ext_vector_type(8))) int          i32x8;
typedef __attribute__((ext_vector_type(4))) int          i32x4;

// 2D bf16 tile DMA: tile0 elems per row (contiguous), tile1 rows,
// row pitch strideElems; written densely to LDS at lds_byte_off.
static __device__ __forceinline__ void tdm_load_2d(
    unsigned lds_byte_off, const unsigned short* gptr,
    unsigned strideElems, unsigned tile0, unsigned tile1)
{
    unsigned long long ga = (unsigned long long)gptr;
    unsigned long long s0 = (unsigned long long)strideElems;
    u32x4 g0;
    g0[0] = 1u;                                               // count=1, user mode
    g0[1] = lds_byte_off;                                     // lds_addr
    g0[2] = (unsigned)(ga & 0xFFFFFFFFu);                     // global_addr[31:0]
    g0[3] = (unsigned)((ga >> 32) & 0x01FFFFFFu) | 0x80000000u; // addr[56:32] | type=2
    i32x8 g1;
    g1[0] = (int)(1u << 16);                                  // data_size=2 bytes
    g1[1] = (int)((tile0 & 0xFFFFu) << 16);                   // tensor_dim0[15:0]
    g1[2] = (int)(((tile0 >> 16) & 0xFFFFu) | ((tile1 & 0xFFFFu) << 16)); // dim0 hi | dim1 lo
    g1[3] = (int)(((tile1 >> 16) & 0xFFFFu) | ((tile0 & 0xFFFFu) << 16)); // dim1 hi | tile_dim0
    g1[4] = (int)(tile1 & 0xFFFFu);                           // tile_dim1 | tile_dim2=0
    g1[5] = (int)(s0 & 0xFFFFFFFFull);                        // dim0_stride[31:0]
    g1[6] = (int)((s0 >> 32) & 0xFFFFull);                    // dim0_stride[47:32] | stride1 lo=0
    g1[7] = 0;
    i32x4 z4 = {0, 0, 0, 0};
#if __clang_major__ >= 23
    i32x8 z8 = {0, 0, 0, 0, 0, 0, 0, 0};
    __builtin_amdgcn_tensor_load_to_lds(g0, g1, z4, z4, z8, 0);
#else
    __builtin_amdgcn_tensor_load_to_lds(g0, g1, z4, z4, 0);
#endif
}
#endif

// -------------------- fp32 -> bf16 conversion --------------------
__global__ void cvt_f32_bf16(const float* __restrict__ src,
                             unsigned short* __restrict__ dst, int n) {
    int i = blockIdx.x * blockDim.x + threadIdx.x;
    int stride = gridDim.x * blockDim.x;
    for (; i < n; i += stride) dst[i] = f2bf(src[i]);
}

// dst[n*K + k] = bf16(src[k*N + n])  (weights pre-transposed so LDS fills are block copies)
__global__ void cvt_transpose_bf16(const float* __restrict__ src,
                                   unsigned short* __restrict__ dst, int K, int N) {
    int k = blockIdx.x * blockDim.x + threadIdx.x;
    int n = blockIdx.y;
    if (k < K) dst[(size_t)n * K + k] = f2bf(src[(size_t)k * N + n]);
}

// -------------------- WMMA GEMM: C = A(bf16)[M,K] * Wt(bf16)[N,K] + bias --------------------
// MODE 0: write fp32 to outF [M,N]
// MODE 1: scatter bf16 into Q/K [B,H,T,D] and V transposed [B,H,D,T]
template<int MODE>
__global__ __launch_bounds__(256) void gemm_bf16(
    const unsigned short* __restrict__ A,    // [M][K]
    const unsigned short* __restrict__ Wt,   // [N][K]  (pre-transposed)
    const float* __restrict__ bias,
    int M, int N, int K,
    float* __restrict__ outF,
    unsigned short* __restrict__ qp,
    unsigned short* __restrict__ kp,
    unsigned short* __restrict__ vp)
{
    __shared__ __attribute__((aligned(16))) unsigned short As[2][128][32]; // [m][k]
    __shared__ __attribute__((aligned(16))) unsigned short Bs[2][64][32];  // [n][k]

    const int tid  = threadIdx.x;
    const int lane = tid & 31;
    const int wave = tid >> 5;
    const int half = lane >> 4;
    const int l16  = lane & 15;
    const int wm   = wave >> 1;    // 0..3 : 32-row strip
    const int wn   = wave & 1;     // 0..1 : 32-col strip

    const int m0 = blockIdx.y * 128;
    const int n0 = blockIdx.x * 64;

    auto loadTile = [&](int bi, int kk) {
#if USE_TDM
        if (wave == 0) {
            tdm_load_2d(lds_off(&As[bi][0][0]), A  + (size_t)m0 * K + kk, (unsigned)K, 32, 128);
            tdm_load_2d(lds_off(&Bs[bi][0][0]), Wt + (size_t)n0 * K + kk, (unsigned)K, 32, 64);
        }
#else
        #pragma unroll
        for (int i = 0; i < 2; ++i) {                  // A: 512 x 16B chunks
            int c   = tid + i * 256;
            int row = c >> 2;
            int kc  = (c & 3) * 8;
            uint4 v = *(const uint4*)(A + (size_t)(m0 + row) * K + kk + kc);
            *(uint4*)(&As[bi][row][kc]) = v;
        }
        {                                              // B: 256 x 16B chunks
            int row = tid >> 2;
            int kc  = (tid & 3) * 8;
            uint4 v = *(const uint4*)(Wt + (size_t)(n0 + row) * K + kk + kc);
            *(uint4*)(&Bs[bi][row][kc]) = v;
        }
#endif
    };

    v8f acc[2][2];
    acc[0][0] = zero8(); acc[0][1] = zero8();
    acc[1][0] = zero8(); acc[1][1] = zero8();

    int buf = 0;
    loadTile(0, 0);
    for (int k0 = 0; k0 < K; k0 += 32) {
        WAIT_COPY();
        __syncthreads();
        if (k0 + 32 < K) loadTile(buf ^ 1, k0 + 32);

        v16bf af[2], bfr[2];
        #pragma unroll
        for (int tm = 0; tm < 2; ++tm) {
            int r = wm * 32 + tm * 16 + l16;
            // A 16x32 bf16 layout: e0..7 -> k=half*8+e ; e8..15 -> k=16+half*8+(e-8)
            af[tm] = make_frag(&As[buf][r][half * 8], &As[buf][r][16 + half * 8]);
        }
        #pragma unroll
        for (int tn = 0; tn < 2; ++tn) {
            int cidx = wn * 32 + tn * 16 + l16;
            // B 32x16 bf16 layout: e0..7 -> k=half*8+e ; e8..15 -> k=16+half*8+(e-8)
            bfr[tn] = make_frag(&Bs[buf][cidx][half * 8], &Bs[buf][cidx][16 + half * 8]);
        }
        #pragma unroll
        for (int tm = 0; tm < 2; ++tm)
            #pragma unroll
            for (int tn = 0; tn < 2; ++tn)
                acc[tm][tn] = __builtin_amdgcn_wmma_f32_16x16x32_bf16(
                    false, af[tm], false, bfr[tn], (short)0, acc[tm][tn], false, false);
        buf ^= 1;
    }

    // epilogue: C/D layout -> lane holds col l16, rows g + 8*half
    #pragma unroll
    for (int tm = 0; tm < 2; ++tm) {
        #pragma unroll
        for (int tn = 0; tn < 2; ++tn) {
            int col = n0 + wn * 32 + tn * 16 + l16;
            float bv = bias[col];
            #pragma unroll
            for (int g = 0; g < 8; ++g) {
                int row = m0 + wm * 32 + tm * 16 + g + half * 8;
                float v = acc[tm][tn][g] + bv;
                if (MODE == 0) {
                    outF[(size_t)row * N + col] = v;
                } else {
                    int sec = col >> 9;          // 0=q 1=k 2=v
                    int h   = (col >> 6) & 7;
                    int d   = col & 63;
                    int b   = row >> 11;
                    int t   = row & (TT - 1);
                    unsigned short bfv = f2bf(v);
                    if (sec == 0)
                        qp[((((size_t)b * HH + h) * TT) + t) * DD + d] = bfv;
                    else if (sec == 1)
                        kp[((((size_t)b * HH + h) * TT) + t) * DD + d] = bfv;
                    else  // V stored transposed [B,H,D,T]
                        vp[((((size_t)b * HH + h) * DD) + d) * TT + t] = bfv;
                }
            }
        }
    }
}

// -------------------- Flash attention --------------------
// grid: (T/64, H, B), block: 128 threads = 4 waves; wave w owns query rows qt*64+w*16 ..+15
__global__ __launch_bounds__(128) void attn_kernel(
    const unsigned short* __restrict__ Qp,   // [B,H,T,D]
    const unsigned short* __restrict__ Kp,   // [B,H,T,D]
    const unsigned short* __restrict__ Vt,   // [B,H,D,T] (transposed)
    unsigned short* __restrict__ Yp)         // [B,T,C] bf16
{
    __shared__ __attribute__((aligned(16))) unsigned short Ks[2][32][64];  // [key][d]
    __shared__ __attribute__((aligned(16))) unsigned short Vs[2][64][32];  // [d][key]
    __shared__ __attribute__((aligned(16))) unsigned short Pw[4][16][32];  // probs bf16
    __shared__ __attribute__((aligned(16))) float Sw[4][16][32];           // raw scores
    __shared__ float rowsc[4][16];

    const int tid  = threadIdx.x;
    const int lane = tid & 31;
    const int wave = tid >> 5;
    const int half = lane >> 4;
    const int l16  = lane & 15;

    const int qt = blockIdx.x;
    const int h  = blockIdx.y;
    const int b  = blockIdx.z;
    const int bh = b * HH + h;
    const int qw = qt * 64 + wave * 16;            // this wave's first query row
    const size_t headBase  = (size_t)bh * TT * DD; // Q/K base
    const size_t headBaseV = (size_t)bh * DD * TT; // Vt base

    auto loadKV = [&](int bi, int kt) {
#if USE_TDM
        if (wave == 0) {
            tdm_load_2d(lds_off(&Ks[bi][0][0]), Kp + headBase + (size_t)(kt * 32) * DD,
                        DD, 64, 32);
            tdm_load_2d(lds_off(&Vs[bi][0][0]), Vt + headBaseV + (size_t)(kt * 32),
                        TT, 32, 64);
        }
#else
        #pragma unroll
        for (int i = 0; i < 2; ++i) {              // K tile: 256 x 16B chunks
            int c   = tid + i * 128;
            int key = c >> 3;
            int dcc = (c & 7) * 8;
            uint4 kv = *(const uint4*)(Kp + headBase + (size_t)(kt * 32 + key) * DD + dcc);
            *(uint4*)(&Ks[bi][key][dcc]) = kv;
        }
        #pragma unroll
        for (int i = 0; i < 2; ++i) {              // V tile: 256 x 16B chunks (already [d][t])
            int c   = tid + i * 128;
            int dr  = c >> 2;
            int kc  = (c & 3) * 8;
            uint4 vv = *(const uint4*)(Vt + headBaseV + (size_t)dr * TT + kt * 32 + kc);
            *(uint4*)(&Vs[bi][dr][kc]) = vv;
        }
#endif
    };

    // preload Q fragments (2 chunks of K=32 over D=64), straight from global
    v16bf qf[2];
    #pragma unroll
    for (int dc = 0; dc < 2; ++dc) {
        const unsigned short* base = Qp + headBase + (size_t)(qw + l16) * DD + dc * 32;
        qf[dc] = make_frag(base + half * 8, base + 16 + half * 8);
    }

    v8f oacc[4];
    #pragma unroll
    for (int nh = 0; nh < 4; ++nh) oacc[nh] = zero8();
    float mrow = NEG_INF;   // meaningful in lanes 0..15 (row = lane)
    float lrow = 0.f;

    const int nkt = qt * 2 + 2;   // causal upper bound for this block (32-key tiles)
    int buf = 0;
    loadKV(0, 0);
    for (int kt = 0; kt < nkt; ++kt) {
        WAIT_COPY();
        __syncthreads();
        if (kt + 1 < nkt) loadKV(buf ^ 1, kt + 1);

        // S = Q * K^T : 16x32 scores via 2 N-halves x 2 D-chunks
        v8f S0 = zero8(), S1 = zero8();
        #pragma unroll
        for (int dc = 0; dc < 2; ++dc) {
            v16bf b0 = make_frag(&Ks[buf][l16][dc * 32 + half * 8],
                                 &Ks[buf][l16][dc * 32 + 16 + half * 8]);
            v16bf b1 = make_frag(&Ks[buf][16 + l16][dc * 32 + half * 8],
                                 &Ks[buf][16 + l16][dc * 32 + 16 + half * 8]);
            S0 = __builtin_amdgcn_wmma_f32_16x16x32_bf16(false, qf[dc], false, b0,
                                                         (short)0, S0, false, false);
            S1 = __builtin_amdgcn_wmma_f32_16x16x32_bf16(false, qf[dc], false, b1,
                                                         (short)0, S1, false, false);
        }
        // spill scores to per-wave LDS for row-wise softmax
        #pragma unroll
        for (int g = 0; g < 8; ++g) {
            Sw[wave][g + half * 8][l16]      = S0[g];
            Sw[wave][g + half * 8][16 + l16] = S1[g];
        }
        asm volatile("s_wait_dscnt 0" ::: "memory");

        // online softmax: lane r<16 owns query row qw+r
        if (lane < 16) {
            int qg = qw + lane;
            float mx = mrow;
            float sv[32];
            #pragma unroll
            for (int j = 0; j < 32; ++j) {
                int kg = kt * 32 + j;
                float s = (kg <= qg) ? Sw[wave][lane][j] * 0.125f : NEG_INF;
                sv[j] = s;
                mx = fmaxf(mx, s);
            }
            float scale = (mx == NEG_INF) ? 1.0f : __expf(mrow - mx);
            float lsum  = lrow * scale;
            #pragma unroll
            for (int j = 0; j < 32; ++j) {
                float p = (sv[j] == NEG_INF) ? 0.0f : __expf(sv[j] - mx);
                lsum += p;
                Pw[wave][lane][j] = f2bf(p);
            }
            mrow = mx;
            lrow = lsum;
            rowsc[wave][lane] = scale;
        }
        asm volatile("s_wait_dscnt 0" ::: "memory");

        // rescale O accumulators by exp(m_old - m_new) per row
        #pragma unroll
        for (int g = 0; g < 8; ++g) {
            float sc = rowsc[wave][g + half * 8];
            #pragma unroll
            for (int nh = 0; nh < 4; ++nh) oacc[nh][g] *= sc;
        }
        // O += P * V
        v16bf pf = make_frag(&Pw[wave][l16][half * 8], &Pw[wave][l16][16 + half * 8]);
        #pragma unroll
        for (int nh = 0; nh < 4; ++nh) {
            v16bf vf = make_frag(&Vs[buf][nh * 16 + l16][half * 8],
                                 &Vs[buf][nh * 16 + l16][16 + half * 8]);
            oacc[nh] = __builtin_amdgcn_wmma_f32_16x16x32_bf16(false, pf, false, vf,
                                                               (short)0, oacc[nh], false, false);
        }
        buf ^= 1;
    }

    // finalize: divide by row sums, emit bf16 y[b, t, h*64 + d]
    if (lane < 16) rowsc[wave][lane] = (lrow > 0.f) ? (1.0f / lrow) : 0.0f;
    asm volatile("s_wait_dscnt 0" ::: "memory");
    #pragma unroll
    for (int g = 0; g < 8; ++g) {
        int m = g + half * 8;
        float inv = rowsc[wave][m];
        int t = qw + m;
        #pragma unroll
        for (int nh = 0; nh < 4; ++nh) {
            int d = nh * 16 + l16;
            Yp[((size_t)(b * TT + t)) * CC + h * DD + d] = f2bf(oacc[nh][g] * inv);
        }
    }
}

// -------------------- host-side orchestration --------------------
extern "C" void kernel_launch(void* const* d_in, const int* in_sizes, int n_in,
                              void* d_out, int out_size, void* d_ws, size_t ws_size,
                              hipStream_t stream) {
    const float* x      = (const float*)d_in[0];
    const float* W_attn = (const float*)d_in[1];
    const float* b_attn = (const float*)d_in[2];
    const float* W_proj = (const float*)d_in[3];
    const float* b_proj = (const float*)d_in[4];
    float* out = (float*)d_out;

    // workspace layout (bf16 buffers), all 256B-aligned offsets
    char* ws = (char*)d_ws;
    unsigned short* xb  = (unsigned short*)(ws);                    // 16 MB : x bf16 [M,K]
    unsigned short* wab = (unsigned short*)(ws + 16777216);         // 1.5 MB: W_attn^T bf16 [N,K]
    unsigned short* wpb = (unsigned short*)(ws + 18350080);         // 0.5 MB: W_proj^T bf16 [N,K]
    unsigned short* qb  = (unsigned short*)(ws + 18874368);         // 16 MB : Q [B,H,T,D]
    unsigned short* kb  = (unsigned short*)(ws + 35651584);         // 16 MB : K [B,H,T,D]
    unsigned short* vb  = (unsigned short*)(ws + 52428800);         // 16 MB : V^T [B,H,D,T]
    unsigned short* yb  = (unsigned short*)(ws + 69206016);         // 16 MB : attn out bf16

    cvt_f32_bf16<<<2048, 256, 0, stream>>>(x, xb, MROWS * CC);
    cvt_transpose_bf16<<<dim3(CC / 256, N_QKV), 256, 0, stream>>>(W_attn, wab, CC, N_QKV);
    cvt_transpose_bf16<<<dim3(CC / 256, CC), 256, 0, stream>>>(W_proj, wpb, CC, CC);

    // QKV projection: [16384,512] x [512,1536] -> scatter Q/K/V bf16
    gemm_bf16<1><<<dim3(N_QKV / 64, MROWS / 128), 256, 0, stream>>>(
        xb, wab, b_attn, MROWS, N_QKV, CC, nullptr, qb, kb, vb);

    // flash attention
    attn_kernel<<<dim3(TT / 64, HH, BB), 128, 0, stream>>>(qb, kb, vb, yb);

    // output projection: [16384,512] x [512,512] -> fp32 out
    gemm_bf16<0><<<dim3(CC / 64, MROWS / 128), 256, 0, stream>>>(
        yb, wpb, b_proj, MROWS, CC, CC, out, nullptr, nullptr, nullptr);
}
